// Transformer_77773267796509
// MI455X (gfx1250) — matrix operand order repaired and checked by reading, not compiled, not run on hardware
//
#include <hip/hip_runtime.h>
#include <hip/hip_bf16.h>

// Problem constants (from reference)
#define WB    64
#define WS    128
#define WLW   16
#define WEC   64
#define WHC   64
#define WEW   300
#define WHW   256
#define WNTAGS 17
#define NSEQ  (WB * WS)          // 8192 word positions == char sequences
#define DWC   (WEW + 2 * WHC)    // 428
#define DWCP  448                // K padded to multiple of 32
#define PADC  84                 // PAD_CHAR

typedef __attribute__((ext_vector_type(16))) __bf16 v16bf;
typedef __attribute__((ext_vector_type(8)))  float  v8f;

union FragU { v16bf v; unsigned int u[8]; };

__device__ __forceinline__ unsigned short f2bf(float x) {
  unsigned int u = __float_as_uint(x);
  u += 0x7FFFu + ((u >> 16) & 1u);   // round-to-nearest-even
  return (unsigned short)(u >> 16);
}
__device__ __forceinline__ float bf2f(unsigned short b) {
  return __uint_as_float(((unsigned int)b) << 16);
}
__device__ __forceinline__ float sigm(float x) {
  return 1.0f / (1.0f + __expf(-x));
}

// A fragment (16x32 bf16, M x K). Lane l holds row M = l%16.
// VGPR j holds K pair: K = (j<4 ? 2j : 8+2j) + (l>=16 ? 8 : 0).
__device__ __forceinline__ v16bf load_a(const unsigned short* src, int stride,
                                        int row0, int kb, int lane) {
  FragU f;
  const unsigned short* p = src + (size_t)(row0 + (lane & 15)) * stride + kb + ((lane >> 4) << 3);
#pragma unroll
  for (int j = 0; j < 8; ++j) {
    int kk = (j < 4) ? (2 * j) : (8 + 2 * j);
    f.u[j] = *(const unsigned int*)(p + kk);
  }
  return f.v;
}

// B fragment (32x16 bf16, K x N) sourced from row-major W[N][K] (B = W^T).
// Lane l holds col N = l%16. VGPR j: K = (j<4 ? 2j : 8+2j) + (l>=16 ? 8 : 0).
__device__ __forceinline__ v16bf load_b(const unsigned short* srcW, int stride,
                                        int col0, int kb, int lane) {
  FragU f;
  const unsigned short* p = srcW + (size_t)(col0 + (lane & 15)) * stride + kb + ((lane >> 4) << 3);
#pragma unroll
  for (int j = 0; j < 8; ++j)
    f.u[j] = *(const unsigned int*)(p + ((j < 4) ? (2 * j) : (8 + 2 * j)));
  return f.v;
}

__device__ __forceinline__ v8f wmma_bf16(v16bf a, v16bf b, v8f c) {
  return __builtin_amdgcn_wmma_f32_16x16x32_bf16(false, a, false, b, (short)0, c,
                                                 false, false);
}

// ---------------------------------------------------------------------------
// f32 -> bf16 conversion with optional K padding (weights)
__global__ void cvt_pad_kernel(const float* __restrict__ src, unsigned short* __restrict__ dst,
                               int rows, int sc, int dc) {
  int total = rows * dc;
  for (int i = blockIdx.x * blockDim.x + threadIdx.x; i < total; i += gridDim.x * blockDim.x) {
    int r = i / dc, c = i - r * dc;
    dst[i] = (c < sc) ? f2bf(src[r * sc + c]) : (unsigned short)0;
  }
}

// Word-embedding gather into bf16 xw[NSEQ][DWCP]: cols [0,300) + zero pad [428,448)
__global__ void word_emb_kernel(const int* __restrict__ words, const float* __restrict__ emb_word,
                                unsigned short* __restrict__ xw) {
  int total = NSEQ * 320;
  for (int i = blockIdx.x * blockDim.x + threadIdx.x; i < total; i += gridDim.x * blockDim.x) {
    int n = i / 320, j = i - n * 320;
    int col = (j < WEW) ? j : (j + 128);      // 300..319 -> 428..447 (zero pad)
    unsigned short v = 0;
    if (j < WEW) v = f2bf(emb_word[(size_t)words[n] * WEW + j]);
    xw[(size_t)n * DWCP + col] = v;
  }
}

// ---------------------------------------------------------------------------
// Char BiLSTM: one block = 16 sequences, one direction (blockIdx.y).
// 128 threads = 4 waves; wave w owns hidden columns [16w,16w+16) for all 4 gates.
// Emits final h into xw columns [300+dir*64, +64).
__global__ void __launch_bounds__(128)
char_lstm_kernel(const int* __restrict__ chars, const float* __restrict__ emb_char,
                 const unsigned short* __restrict__ Wih_f, const unsigned short* __restrict__ Whh_f,
                 const float* __restrict__ b_f,
                 const unsigned short* __restrict__ Wih_b, const unsigned short* __restrict__ Whh_b,
                 const float* __restrict__ b_b,
                 unsigned short* __restrict__ xw) {
  extern __shared__ char smem[];
  unsigned short* s_emb = (unsigned short*)smem;        // [16 seq][16 t][64]
  unsigned short* s_wih = s_emb + 16 * 16 * 64;         // [256][64] bf16
  unsigned short* s_whh = s_wih + 256 * 64;             // [256][64] bf16
  unsigned short* s_h   = s_whh + 256 * 64;             // [16][64] bf16
  float* s_c    = (float*)(s_h + 16 * 64);              // [16][64] f32
  float* s_bias = s_c + 16 * 64;                        // [256]
  int*   s_len  = (int*)(s_bias + 256);                 // [16]

  const int tid = threadIdx.x;
  const int dir = blockIdx.y;
  const int seq0 = blockIdx.x * 16;
  const unsigned short* Wih = dir ? Wih_b : Wih_f;
  const unsigned short* Whh = dir ? Whh_b : Whh_f;
  const float* bias = dir ? b_b : b_f;

  for (int i = tid; i < 256 * 64; i += 128) { s_wih[i] = Wih[i]; s_whh[i] = Whh[i]; }
  for (int i = tid; i < 256; i += 128) s_bias[i] = bias[i];
  if (tid < 16) {
    int len = 0;
    for (int t = 0; t < WLW; ++t) len += (chars[(seq0 + tid) * WLW + t] != PADC) ? 1 : 0;
    s_len[tid] = len;
  }
  for (int i = tid; i < 16 * 16 * 64; i += 128) {
    int ec = i & 63, t = (i >> 6) & 15, sq = i >> 10;
    int ch = chars[(seq0 + sq) * WLW + t];
    s_emb[i] = f2bf(emb_char[(size_t)ch * WEC + ec]);
  }
  for (int i = tid; i < 16 * 64; i += 128) { s_h[i] = 0; s_c[i] = 0.0f; }
  __syncthreads();

  const int lane = tid & 31, wave = tid >> 5;
  const int colL = lane & 15;
  for (int step = 0; step < WLW; ++step) {
    int t = dir ? (WLW - 1 - step) : step;
    v16bf ax0 = load_a(s_emb + t * 64, WLW * 64, 0, 0,  lane);
    v16bf ax1 = load_a(s_emb + t * 64, WLW * 64, 0, 32, lane);
    v16bf ah0 = load_a(s_h, 64, 0, 0,  lane);
    v16bf ah1 = load_a(s_h, 64, 0, 32, lane);
    __syncthreads();                 // all h reads complete before any h write

    v8f acc[4];
#pragma unroll
    for (int gi = 0; gi < 4; ++gi) {
      int gcol = gi * 64 + wave * 16;
      float bv = s_bias[gcol + colL];
      v8f c;
#pragma unroll
      for (int r = 0; r < 8; ++r) c[r] = bv;
      c = wmma_bf16(ax0, load_b(s_wih, 64, gcol, 0,  lane), c);
      c = wmma_bf16(ax1, load_b(s_wih, 64, gcol, 32, lane), c);
      c = wmma_bf16(ah0, load_b(s_whh, 64, gcol, 0,  lane), c);
      c = wmma_bf16(ah1, load_b(s_whh, 64, gcol, 32, lane), c);
      acc[gi] = c;
    }
    int col = wave * 16 + colL;
#pragma unroll
    for (int r = 0; r < 8; ++r) {
      int row = r + ((lane >> 4) << 3);
      float iv = sigm(acc[0][r]);
      float fv = sigm(acc[1][r]);
      float gv = tanhf(acc[2][r]);
      float ov = sigm(acc[3][r]);
      float cold = s_c[row * 64 + col];
      float cn = fv * cold + iv * gv;
      float hn = ov * tanhf(cn);
      int len = s_len[row]; int leneff = (len > 0) ? len : 1;   // jnp.maximum(lens,1)
      bool valid = t < leneff;
      float hold = bf2f(s_h[row * 64 + col]);
      s_c[row * 64 + col] = valid ? cn : cold;
      s_h[row * 64 + col] = f2bf(valid ? hn : hold);
    }
    __syncthreads();
  }

  const int colbase = 300 + dir * 64;
  int col = wave * 16 + colL;
#pragma unroll
  for (int r = 0; r < 8; ++r) {
    int row = r + ((lane >> 4) << 3);
    int n = seq0 + row;
    xw[(size_t)n * DWCP + colbase + col] =
        (s_len[row] > 0) ? s_h[row * 64 + col] : (unsigned short)0;   // * (len>0) mask
  }
}

// ---------------------------------------------------------------------------
// xg = xw @ Ww_ih^T + b  for both directions. Output scattered to [t][b][1024].
__global__ void __launch_bounds__(256)
xg_gemm_kernel(const unsigned short* __restrict__ xw,
               const unsigned short* __restrict__ Wf, const float* __restrict__ bfp,
               const unsigned short* __restrict__ Wb, const float* __restrict__ bbp,
               float* __restrict__ xgf, float* __restrict__ xgb) {
  const int lane = threadIdx.x & 31, wave = threadIdx.x >> 5;
  const int dir = blockIdx.z;
  const unsigned short* W = dir ? Wb : Wf;
  const float* bias = dir ? bbp : bfp;
  float* xg = dir ? xgb : xgf;
  const int m0 = blockIdx.x * 16;
  const int col0 = blockIdx.y * 128 + wave * 16;

  v8f c;
  float bv = bias[col0 + (lane & 15)];
#pragma unroll
  for (int r = 0; r < 8; ++r) c[r] = bv;
  for (int kb = 0; kb < DWCP; kb += 32) {
    v16bf a = load_a(xw, DWCP, m0, kb, lane);
    v16bf b = load_b(W, DWCP, col0, kb, lane);
    c = wmma_bf16(a, b, c);
  }
  int col = col0 + (lane & 15);
#pragma unroll
  for (int r = 0; r < 8; ++r) {
    int n = m0 + r + ((lane >> 4) << 3);
    int b_ = n >> 7, t = n & 127;                       // n = b*128 + t
    xg[(size_t)(t * WB + b_) * 1024 + col] = c[r];
  }
}

// ---------------------------------------------------------------------------
// Word LSTM recurrence. One block per direction, 1024 threads = 32 waves.
// h double-buffered bf16 in LDS, c f32 in LDS, Whh streamed from L2.
// 64 tasks/step = (4 row tiles) x (16 hidden col tiles); 2 tasks/wave;
// each task computes i,f,g,o 16x16 tiles over K=256 (32 WMMAs) and fuses the
// elementwise LSTM cell update + masked output store.
__global__ void __launch_bounds__(1024)
word_lstm_kernel(const float* __restrict__ xgf, const float* __restrict__ xgb,
                 const unsigned short* __restrict__ Whhf, const unsigned short* __restrict__ Whhb,
                 const int* __restrict__ lens, float* __restrict__ enc) {
  extern __shared__ char smem[];
  unsigned short* s_h0 = (unsigned short*)smem;          // [64][256] bf16
  unsigned short* s_h1 = s_h0 + 64 * 256;                // [64][256] bf16
  float* s_c  = (float*)(s_h1 + 64 * 256);               // [64][256] f32
  int*   s_len = (int*)(s_c + 64 * 256);                 // [64]

  const int tid = threadIdx.x;
  const int dir = blockIdx.x;
  const float* xg = dir ? xgb : xgf;
  const unsigned short* Whh = dir ? Whhb : Whhf;
  const int outcol = dir * WHW;

  for (int i = tid; i < 64 * 256; i += 1024) { s_h0[i] = 0; s_h1[i] = 0; s_c[i] = 0.0f; }
  if (tid < 64) s_len[tid] = lens[tid];
  __syncthreads();

  const int lane = tid & 31, wave = tid >> 5;
  for (int step = 0; step < WS; ++step) {
    const int t = dir ? (WS - 1 - step) : step;
    const unsigned short* hcur = (step & 1) ? s_h1 : s_h0;
    unsigned short* hnext = (step & 1) ? s_h0 : s_h1;
#pragma unroll
    for (int task_i = 0; task_i < 2; ++task_i) {
      int task = wave + task_i * 32;
      int rt = task >> 4;        // row tile (batch rows rt*16..)
      int ht = task & 15;        // hidden col tile
      v8f acc[4];
#pragma unroll
      for (int gi = 0; gi < 4; ++gi) {
        int gcol = gi * WHW + ht * 16 + (lane & 15);
#pragma unroll
        for (int r = 0; r < 8; ++r) {
          int b_ = rt * 16 + r + ((lane >> 4) << 3);
          acc[gi][r] = xg[(size_t)(t * WB + b_) * 1024 + gcol];
        }
      }
      for (int kb = 0; kb < WHW; kb += 32) {
        v16bf a = load_a(hcur, WHW, rt * 16, kb, lane);
#pragma unroll
        for (int gi = 0; gi < 4; ++gi) {
          v16bf b = load_b(Whh, WHW, gi * WHW + ht * 16, kb, lane);
          acc[gi] = wmma_bf16(a, b, acc[gi]);
        }
      }
      int col = ht * 16 + (lane & 15);
#pragma unroll
      for (int r = 0; r < 8; ++r) {
        int b_ = rt * 16 + r + ((lane >> 4) << 3);
        float iv = sigm(acc[0][r]);
        float fv = sigm(acc[1][r]);
        float gv = tanhf(acc[2][r]);
        float ov = sigm(acc[3][r]);
        float cold = s_c[b_ * WHW + col];
        float cn = fv * cold + iv * gv;
        float hn = ov * tanhf(cn);
        bool valid = t < s_len[b_];
        s_c[b_ * WHW + col] = valid ? cn : cold;
        float hold = bf2f(hcur[b_ * WHW + col]);
        hnext[b_ * WHW + col] = f2bf(valid ? hn : hold);
        enc[(size_t)(b_ * WS + t) * (2 * WHW) + outcol + col] = valid ? hn : 0.0f;
      }
    }
    __syncthreads();   // writes to hnext visible; next step reads it as hcur
  }
}

// ---------------------------------------------------------------------------
// Tag projection: out[n][tg] = enc[n][:] . W_tag[tg][:] + b_tag[tg]   (tiny)
__global__ void tag_kernel(const float* __restrict__ enc, const float* __restrict__ Wt,
                           const float* __restrict__ bt, float* __restrict__ out) {
  int i = blockIdx.x * blockDim.x + threadIdx.x;
  if (i >= NSEQ * WNTAGS) return;
  int n = i / WNTAGS, tg = i - n * WNTAGS;
  float s = bt[tg];
  const float* e = enc + (size_t)n * 512;
  const float* w = Wt + (size_t)tg * 512;
  for (int k = 0; k < 512; ++k) s += e[k] * w[k];
  out[i] = s;
}

// ---------------------------------------------------------------------------
extern "C" void kernel_launch(void* const* d_in, const int* in_sizes, int n_in,
                              void* d_out, int out_size, void* d_ws, size_t ws_size,
                              hipStream_t stream) {
  (void)in_sizes; (void)n_in; (void)out_size; (void)ws_size;
  const int*   words     = (const int*)d_in[0];
  const int*   chars     = (const int*)d_in[1];
  const int*   lens_word = (const int*)d_in[2];
  const float* emb_char  = (const float*)d_in[3];
  const float* emb_word  = (const float*)d_in[4];
  const float* Wc_ih_f   = (const float*)d_in[5];
  const float* Wc_hh_f   = (const float*)d_in[6];
  const float* bc_f      = (const float*)d_in[7];
  const float* Wc_ih_b   = (const float*)d_in[8];
  const float* Wc_hh_b   = (const float*)d_in[9];
  const float* bc_b      = (const float*)d_in[10];
  const float* Ww_ih_f   = (const float*)d_in[11];
  const float* Ww_hh_f   = (const float*)d_in[12];
  const float* bw_f      = (const float*)d_in[13];
  const float* Ww_ih_b   = (const float*)d_in[14];
  const float* Ww_hh_b   = (const float*)d_in[15];
  const float* bw_b      = (const float*)d_in[16];
  const float* W_tag     = (const float*)d_in[17];
  const float* b_tag     = (const float*)d_in[18];
  float* out = (float*)d_out;

  char* ws = (char*)d_ws;
  size_t off = 0;
  auto take = [&](size_t bytes) -> char* {
    char* p = ws + off;
    off = (off + bytes + 255) & ~(size_t)255;
    return p;
  };
  unsigned short* wcih_f16[2];
  unsigned short* wchh_f16[2];
  unsigned short* wwih_f16[2];
  unsigned short* wwhh_f16[2];
  for (int d = 0; d < 2; ++d) {
    wcih_f16[d] = (unsigned short*)take(256 * 64 * 2);
    wchh_f16[d] = (unsigned short*)take(256 * 64 * 2);
    wwih_f16[d] = (unsigned short*)take((size_t)1024 * DWCP * 2);
    wwhh_f16[d] = (unsigned short*)take((size_t)1024 * WHW * 2);
  }
  unsigned short* xw = (unsigned short*)take((size_t)NSEQ * DWCP * 2);
  float* xg_f = (float*)take((size_t)WS * WB * 1024 * 4);
  float* xg_b = (float*)take((size_t)WS * WB * 1024 * 4);
  float* enc  = (float*)take((size_t)NSEQ * 512 * 4);

  auto grid1 = [](int total) { return dim3((unsigned)((total + 255) / 256)); };

  // 1) weight conversions to bf16 (with K padding for Ww_ih)
  cvt_pad_kernel<<<grid1(256 * 64), 256, 0, stream>>>(Wc_ih_f, wcih_f16[0], 256, 64, 64);
  cvt_pad_kernel<<<grid1(256 * 64), 256, 0, stream>>>(Wc_hh_f, wchh_f16[0], 256, 64, 64);
  cvt_pad_kernel<<<grid1(256 * 64), 256, 0, stream>>>(Wc_ih_b, wcih_f16[1], 256, 64, 64);
  cvt_pad_kernel<<<grid1(256 * 64), 256, 0, stream>>>(Wc_hh_b, wchh_f16[1], 256, 64, 64);
  cvt_pad_kernel<<<grid1(1024 * DWCP), 256, 0, stream>>>(Ww_ih_f, wwih_f16[0], 1024, DWC, DWCP);
  cvt_pad_kernel<<<grid1(1024 * DWCP), 256, 0, stream>>>(Ww_ih_b, wwih_f16[1], 1024, DWC, DWCP);
  cvt_pad_kernel<<<grid1(1024 * WHW), 256, 0, stream>>>(Ww_hh_f, wwhh_f16[0], 1024, WHW, WHW);
  cvt_pad_kernel<<<grid1(1024 * WHW), 256, 0, stream>>>(Ww_hh_b, wwhh_f16[1], 1024, WHW, WHW);

  // 2) word embedding gather (cols 0..299 + zero pad 428..447 of xw)
  word_emb_kernel<<<grid1(NSEQ * 320), 256, 0, stream>>>(words, emb_word, xw);

  // 3) char BiLSTM -> xw cols 300..427 (both directions in one grid)
  const size_t CHAR_LDS =
      (size_t)(16 * 16 * 64 + 256 * 64 + 256 * 64 + 16 * 64) * 2 +
      (size_t)(16 * 64 + 256) * 4 + 16 * 4;   // = 105536 bytes
  char_lstm_kernel<<<dim3(NSEQ / 16, 2), 128, CHAR_LDS, stream>>>(
      chars, emb_char, wcih_f16[0], wchh_f16[0], bc_f,
      wcih_f16[1], wchh_f16[1], bc_b, xw);

  // 4) word input-gate GEMM (both directions)
  xg_gemm_kernel<<<dim3(NSEQ / 16, 8, 2), 256, 0, stream>>>(
      xw, wwih_f16[0], bw_f, wwih_f16[1], bw_b, xg_f, xg_b);

  // 5) word BiLSTM recurrence (dir = blockIdx.x)
  const size_t WORD_LDS = (size_t)(64 * 256) * 2 * 2 + (size_t)(64 * 256) * 4 + 64 * 4; // 131328
  word_lstm_kernel<<<dim3(2), 1024, WORD_LDS, stream>>>(
      xg_f, xg_b, wwhh_f16[0], wwhh_f16[1], lens_word, enc);

  // 6) tag projection
  tag_kernel<<<grid1(NSEQ * WNTAGS), 256, 0, stream>>>(enc, W_tag, b_tag, out);
}
